// DeepBSDE_SABR_11802570130342
// MI455X (gfx1250) — matrix operand order repaired and checked by reading, not compiled, not run on hardware
//
#include <hip/hip_runtime.h>
#include <hip/hip_bf16.h>
#include <math.h>

// ---------------- problem constants ----------------
#define NSTEP   50
#define BATCH   65536
#define HID     64
#define TPB     256
#define NBLK    (BATCH / TPB)          // 256 persistent blocks, 1 row per thread

#define RHO_    (-0.5f)
#define RHOC_   (0.86602540378443864676f)   // sqrt(1-RHO^2)
#define NU_     (0.4f)
#define SQRTDT_ (0.14142135623730950488f)   // sqrt(T/N)
#define FMIN_   80.0f
#define FSPAN_  40.0f
#define AMIN_   0.1f
#define ASPAN_  0.4f
#define KSTR_   100.0f
#define BETA_   0.7f
#define BNEPS_  1e-5f
#define INVB_   (1.0f / (float)BATCH)

// d_ws accumulator layout (floats per ping-pong buffer):
//   [0..4]    BN1 stats (sumF,sumA,sumF2,sumA2,sumFA)
//   [8..71]   BN2 column sums
//   [72..135] BN2 column sum-of-squares
#define ACC_N      136
#define BUF_STRIDE 144

typedef __attribute__((ext_vector_type(16))) _Float16 v16h;
typedef __attribute__((ext_vector_type(8)))  float    v8f;
typedef __attribute__((ext_vector_type(4)))  unsigned v4u_t;
typedef __attribute__((ext_vector_type(8)))  int      v8i_t;
typedef __attribute__((ext_vector_type(4)))  int      v4i_t;

#if __has_builtin(__builtin_amdgcn_tensor_load_to_lds)
#define HAVE_TDM 1
#else
#define HAVE_TDM 0
#endif

#if HAVE_TDM
// TDM D# for a dense 2-D 64x64 f32 tile: global (row-major, stride 64) -> LDS.
// Group0: count=1, lds_addr, global_addr[56:0], type=2.
// Group1: data_size=4B, tensor_dim0/1=64, tile_dim0/1=64, dim0_stride=64.
// This toolchain exposes the 6-arg builtin: (g0, g1, g2, g3, g4, cpol).
__device__ __forceinline__ void tdm_load_64x64_f32(const float* gsrc, unsigned ldsByteOff) {
  unsigned long long ga = (unsigned long long)(__SIZE_TYPE__)gsrc;
  v4u_t g0;
  g0[0] = 1u;                                             // count=1, user descriptor
  g0[1] = ldsByteOff;                                     // lds_addr
  g0[2] = (unsigned)(ga & 0xffffffffull);                 // global_addr[31:0]
  g0[3] = (unsigned)((ga >> 32) & 0x01ffffffull) | 0x80000000u; // addr[56:32] | type=2
  v8i_t g1;
  g1[0] = (int)(2u << 16);        // workgroup_mask=0, data_size=2 (4 bytes)
  g1[1] = (int)(64u << 16);       // tensor_dim0[15:0]=64 (bits 63:48)
  g1[2] = (int)(64u << 16);       // tensor_dim1[15:0]=64 (bits 95:80)
  g1[3] = (int)(64u << 16);       // tile_dim0=64        (bits 127:112)
  g1[4] = (int)64;                // tile_dim1=64, tile_dim2=0
  g1[5] = (int)64;                // tensor_dim0_stride=64
  g1[6] = (int)(4096u << 16);     // tensor_dim1_stride[15:0]=4096 (unused for 2D tile)
  g1[7] = 0;
  v4i_t z4 = {0, 0, 0, 0};        // groups 2/3 unused (<=2D tensor)
  v8i_t z8 = {0, 0, 0, 0, 0, 0, 0, 0};
  __builtin_amdgcn_tensor_load_to_lds(g0, g1, z4, z4, z8, 0);
}
#endif

__device__ __forceinline__ float gld_agent(const float* p) {
  return __hip_atomic_load(p, __ATOMIC_RELAXED, __HIP_MEMORY_SCOPE_AGENT);
}

// Device-wide split barrier. Last arriver re-zeroes the *other* ping-pong
// accumulator buffer (safe: its readers finished before arriving here, its
// next writers start only after release).
__device__ __forceinline__ void grid_barrier(unsigned* bar, float* bufs, int phase) {
  __syncthreads();
  if (threadIdx.x == 0) {
    __threadfence();
    unsigned gen  = __hip_atomic_load(&bar[1], __ATOMIC_ACQUIRE, __HIP_MEMORY_SCOPE_AGENT);
    unsigned prev = __hip_atomic_fetch_add(&bar[0], 1u, __ATOMIC_ACQ_REL, __HIP_MEMORY_SCOPE_AGENT);
    if (prev == gridDim.x - 1u) {
      float* zb = bufs + ((unsigned)(phase + 1) & 1u) * BUF_STRIDE;
      for (int i = 0; i < ACC_N; ++i) zb[i] = 0.0f;
      __threadfence();
      __hip_atomic_store(&bar[0], 0u, __ATOMIC_RELEASE, __HIP_MEMORY_SCOPE_AGENT);
      __hip_atomic_fetch_add(&bar[1], 1u, __ATOMIC_RELEASE, __HIP_MEMORY_SCOPE_AGENT);
    } else {
      while (__hip_atomic_load(&bar[1], __ATOMIC_ACQUIRE, __HIP_MEMORY_SCOPE_AGENT) == gen)
        __builtin_amdgcn_s_sleep(1);
    }
    __threadfence();
  }
  __syncthreads();
}

struct KParams {
  const float* eps;
  const float* F0u;
  const float* A0u;
  const float* w[30];   // u0[0..9], z0[10..19], zs[20..29]
  float*       out;
  unsigned*    bar;
  float*       bufs;
};

__global__ void init_ws(unsigned* bar, float* bufs) {
  if (threadIdx.x < 2) bar[threadIdx.x] = 0u;
  for (int i = threadIdx.x; i < 2 * BUF_STRIDE; i += blockDim.x) bufs[i] = 0.0f;
}

__global__ void __launch_bounds__(TPB, 1)
sabr_persistent(KParams p) {
  __shared__ __align__(16) _Float16 aLDS[TPB * HID];   // activations (WMMA A, row-major f16)
  __shared__ __align__(16) _Float16 w2T[HID * HID];    // W2^T in WMMA-B layout (f16)
  __shared__ __align__(16) float    w2raw[HID * HID];  // TDM staging (raw f32 W2)
  __shared__ float sc1a[HID], sc1b[HID], sh1[HID];
  __shared__ float sc2[HID], sh2[HID];
  __shared__ float colsum[HID], colsq[HID];
  __shared__ float bn5[5];
  __shared__ float b2s[HID], w3s[2 * HID], b3s[2];

  const int tid  = threadIdx.x;
  const int lane = tid & 31;
  const int wid  = tid >> 5;
  const int hi   = lane >> 4;     // lane half (wave32 WMMA layouts)
  const int l15  = lane & 15;
  const int r    = blockIdx.x * TPB + tid;

  int phase = 0;
  const unsigned w2rawOff = (unsigned)(__SIZE_TYPE__)(const void*)&w2raw[0]; // flat LDS addr low32 = LDS offset

  float F = FMIN_ + FSPAN_ * p.F0u[r];
  float a = AMIN_ + ASPAN_ * p.A0u[r];

#if HAVE_TDM
  if (wid == 0) tdm_load_64x64_f32(p.w[4], w2rawOff);   // prime pipeline with u0_W2
#endif

  float y = 0.f, z0 = 0.f, z1 = 0.f;

  auto mlp_eval = [&](const float* W1, const float* b1, const float* g1, const float* be1,
                      const float* W2cur, const float* b2, const float* g2, const float* be2,
                      const float* W3, const float* b3, const float* W2next,
                      int nout, float& o0, float& o1) {
    (void)W2cur;
    // ---- BN1: layer-1 is rank-2 in (F,a); batch stats need only 5 scalars ----
    if (tid < 5) bn5[tid] = 0.f;
    __syncthreads();
    {
      float v0 = F, v1 = a, v2 = F * F, v3 = a * a, v4 = F * a;
      #pragma unroll
      for (int off = 16; off; off >>= 1) {
        v0 += __shfl_xor(v0, off); v1 += __shfl_xor(v1, off); v2 += __shfl_xor(v2, off);
        v3 += __shfl_xor(v3, off); v4 += __shfl_xor(v4, off);
      }
      if (lane == 0) {
        atomicAdd(&bn5[0], v0); atomicAdd(&bn5[1], v1); atomicAdd(&bn5[2], v2);
        atomicAdd(&bn5[3], v3); atomicAdd(&bn5[4], v4);
      }
    }
    __syncthreads();
    float* gb = p.bufs + (phase & 1) * BUF_STRIDE;
    if (tid < 5) atomicAdd(&gb[tid], bn5[tid]);
    grid_barrier(p.bar, p.bufs, phase);
    ++phase;
    if (tid < HID) {
      float mF = gld_agent(&gb[0]) * INVB_, mA = gld_agent(&gb[1]) * INVB_;
      float vF = gld_agent(&gb[2]) * INVB_ - mF * mF;
      float vA = gld_agent(&gb[3]) * INVB_ - mA * mA;
      float cFA = gld_agent(&gb[4]) * INVB_ - mF * mA;
      float w0 = W1[2 * tid], w1 = W1[2 * tid + 1];
      float var = vF * w0 * w0 + vA * w1 * w1 + 2.f * cFA * w0 * w1;
      float s = g1[tid] * rsqrtf(var + BNEPS_);
      sc1a[tid] = w0 * s;
      sc1b[tid] = w1 * s;
      sh1[tid]  = be1[tid] - (mF * w0 + mA * w1) * s;   // b1 cancels in (x-mean)
      b2s[tid]  = b2[tid];
    }
    if (tid < 2 * HID) w3s[tid] = (tid < nout * HID) ? W3[tid] : 0.f;
    if (tid < 2)       b3s[tid] = (tid < nout) ? b3[tid] : 0.f;
    __syncthreads();

    // ---- layer-1 activations -> LDS (f16 row-major: matches WMMA A addressing) ----
    #pragma unroll
    for (int j = 0; j < HID; j += 2) {
      float h0 = fmaxf(0.f, fmaf(F, sc1a[j],     fmaf(a, sc1b[j],     sh1[j])));
      float h1 = fmaxf(0.f, fmaf(F, sc1a[j + 1], fmaf(a, sc1b[j + 1], sh1[j + 1])));
      union { _Float16 h[2]; unsigned u; } pk;
      pk.h[0] = (_Float16)h0; pk.h[1] = (_Float16)h1;
      *(unsigned*)&aLDS[tid * HID + j] = pk.u;
    }

    // ---- W2 arrived via Tensor Data Mover; transpose+convert to f16 B operand ----
#if HAVE_TDM
    if (wid == 0) __builtin_amdgcn_s_wait_tensorcnt((short)0);
    __syncthreads();
    for (int i = tid; i < HID * HID; i += TPB) {
      int k = i & (HID - 1), n = i >> 6;                // contiguous LDS reads per wave
      w2T[n * HID + k] = (_Float16)w2raw[n * HID + k];  // B[k][n] = W2[n][k]
    }
    __syncthreads();
    if (W2next && wid == 0) tdm_load_64x64_f32(W2next, w2rawOff);  // async prefetch next W2
#else
    __syncthreads();
    for (int i = tid; i < HID * HID; i += TPB) {
      int k = i & (HID - 1), n = i >> 6;
      w2T[n * HID + k] = (_Float16)W2cur[n * HID + k];
    }
    __syncthreads();
#endif

    // ---- layer-2 GEMM (256x64)*(64x64) on v_wmma_f32_16x16x32_f16 ----
    v8f acc[2][4];
    #pragma unroll
    for (int mt = 0; mt < 2; ++mt)
      #pragma unroll
      for (int nt = 0; nt < 4; ++nt) {
        v8f zz = {0.f, 0.f, 0.f, 0.f, 0.f, 0.f, 0.f, 0.f};
        acc[mt][nt] = zz;
      }

    // A fragments: lane l (m=l&15) holds K = kc*32 + g*16 + hi*8 + [0..7] per 16B chunk
    v16h afrag[2][2];
    #pragma unroll
    for (int mt = 0; mt < 2; ++mt) {
      int row = (wid * 2 + mt) * 16 + l15;
      #pragma unroll
      for (int kc = 0; kc < 2; ++kc) {
        union { uint4 q[2]; v16h h; } u;
        u.q[0] = *(const uint4*)&aLDS[row * HID + kc * 32 + hi * 8];
        u.q[1] = *(const uint4*)&aLDS[row * HID + kc * 32 + 16 + hi * 8];
        afrag[mt][kc] = u.h;
      }
    }
    #pragma unroll
    for (int nt = 0; nt < 4; ++nt) {
      #pragma unroll
      for (int kc = 0; kc < 2; ++kc) {
        // B fragment: lane l holds column N = nt*16 + l15, K = kc*32 + g*16 + hi*8 + [0..7] per 16B chunk
        int kk = nt * 16 + l15;
        union { uint4 q[2]; v16h h; } ub;
        ub.q[0] = *(const uint4*)&w2T[kk * HID + kc * 32 + hi * 8];
        ub.q[1] = *(const uint4*)&w2T[kk * HID + kc * 32 + 16 + hi * 8];
        #pragma unroll
        for (int mt = 0; mt < 2; ++mt)
          acc[mt][nt] = __builtin_amdgcn_wmma_f32_16x16x32_f16(
              false, afrag[mt][kc], false, ub.h, (short)0, acc[mt][nt], false, false);
      }
    }

    // ---- BN2: 64 column sums + sum-of-squares over the whole batch ----
    if (tid < HID) { colsum[tid] = 0.f; colsq[tid] = 0.f; }
    __syncthreads();
    #pragma unroll
    for (int nt = 0; nt < 4; ++nt) {
      int col = nt * 16 + l15;
      float b2v = b2s[col];
      float s1 = 0.f, s2 = 0.f;
      #pragma unroll
      for (int mt = 0; mt < 2; ++mt)
        #pragma unroll
        for (int i = 0; i < 8; ++i) {
          float v = acc[mt][nt][i] + b2v;
          acc[mt][nt][i] = v;
          s1 += v; s2 += v * v;
        }
      atomicAdd(&colsum[col], s1);
      atomicAdd(&colsq[col], s2);
    }
    __syncthreads();
    gb = p.bufs + (phase & 1) * BUF_STRIDE;
    if (tid < HID) {
      atomicAdd(&gb[8 + tid], colsum[tid]);
      atomicAdd(&gb[72 + tid], colsq[tid]);
    }
    grid_barrier(p.bar, p.bufs, phase);
    ++phase;
    if (tid < HID) {
      float mean = gld_agent(&gb[8 + tid]) * INVB_;
      float var  = gld_agent(&gb[72 + tid]) * INVB_ - mean * mean;
      float s = g2[tid] * rsqrtf(var + BNEPS_);
      sc2[tid] = s;
      sh2[tid] = be2[tid] - mean * s;
    }
    __syncthreads();

    // ---- apply BN2+ReLU, spill h2 (f16) and do the tiny 64->nout layer 3 ----
    #pragma unroll
    for (int nt = 0; nt < 4; ++nt) {
      int col = nt * 16 + l15;
      float s = sc2[col], sh = sh2[col];
      #pragma unroll
      for (int mt = 0; mt < 2; ++mt) {
        int rbase = (wid * 2 + mt) * 16 + hi * 8;       // C/D layout: VGPR i -> M=i+8*hi
        #pragma unroll
        for (int i = 0; i < 8; ++i) {
          float h = fmaxf(0.f, fmaf(acc[mt][nt][i], s, sh));
          aLDS[(rbase + i) * HID + col] = (_Float16)h;
        }
      }
    }
    __syncthreads();

    float r0 = b3s[0], r1 = b3s[1];
    #pragma unroll 16
    for (int j = 0; j < HID; ++j) {
      float h = (float)aLDS[tid * HID + j];
      r0 = fmaf(h, w3s[j], r0);
      r1 = fmaf(h, w3s[HID + j], r1);
    }
    o0 = r0;
    o1 = (nout > 1) ? r1 : 0.f;
    __syncthreads();
  };

  // y = MLP_u0(x0)[:,0]
  float t0, t1;
  mlp_eval(p.w[0], p.w[1], p.w[2], p.w[3], p.w[4], p.w[5], p.w[6], p.w[7], p.w[8], p.w[9],
           p.w[14], 1, t0, t1);
  y = t0;
  // z = MLP_z0(x0)
  mlp_eval(p.w[10], p.w[11], p.w[12], p.w[13], p.w[14], p.w[15], p.w[16], p.w[17], p.w[18], p.w[19],
           p.w[24], 2, z0, z1);

  const float* eRow = p.eps + (size_t)r * (2 * NSTEP);
  for (int n = 0; n < NSTEP - 1; ++n) {
    float e0 = eRow[n];
    float e1 = eRow[NSTEP + n];
    float dW = e0 * SQRTDT_;
    float dZ = (RHO_ * e0 + RHOC_ * e1) * SQRTDT_;
    y = fmaf(z0, dW, fmaf(z1, dZ, y));
    F = fmaf(a * powf(F, BETA_), dW, F);
    a = fmaf(NU_ * a, dZ, a);
    const float* W2next = (n + 1 < NSTEP - 1) ? (p.w[24] + (size_t)(n + 1) * HID * HID) : nullptr;
    if (tid == 0 && n + 1 < NSTEP - 1) {                // warm next step's small weights
      __builtin_prefetch(p.w[20] + (size_t)(n + 1) * 2 * HID, 0, 1);
      __builtin_prefetch(p.w[25] + (size_t)(n + 1) * HID, 0, 1);
    }
    mlp_eval(p.w[20] + (size_t)n * 2 * HID,   p.w[21] + (size_t)n * HID,
             p.w[22] + (size_t)n * HID,       p.w[23] + (size_t)n * HID,
             p.w[24] + (size_t)n * HID * HID, p.w[25] + (size_t)n * HID,
             p.w[26] + (size_t)n * HID,       p.w[27] + (size_t)n * HID,
             p.w[28] + (size_t)n * 2 * HID,   p.w[29] + (size_t)n * 2,
             W2next, 2, z0, z1);
  }
  {
    float e0 = eRow[NSTEP - 1];
    float e1 = eRow[2 * NSTEP - 1];
    float dW = e0 * SQRTDT_;
    float dZ = (RHO_ * e0 + RHOC_ * e1) * SQRTDT_;
    y = fmaf(z0, dW, fmaf(z1, dZ, y));
    F = fmaf(a * powf(F, BETA_), dW, F);
  }
  p.out[r] = y - fmaxf(F - KSTR_, 0.f);
}

extern "C" void kernel_launch(void* const* d_in, const int* in_sizes, int n_in,
                              void* d_out, int out_size, void* d_ws, size_t ws_size,
                              hipStream_t stream) {
  (void)in_sizes; (void)n_in; (void)out_size; (void)ws_size;
  KParams p;
  p.eps = (const float*)d_in[0];
  p.F0u = (const float*)d_in[1];
  p.A0u = (const float*)d_in[2];
  for (int i = 0; i < 30; ++i) p.w[i] = (const float*)d_in[3 + i];
  p.out  = (float*)d_out;
  p.bar  = (unsigned*)d_ws;
  p.bufs = (float*)d_ws + 8;

  init_ws<<<1, 256, 0, stream>>>(p.bar, p.bufs);        // ws is poisoned: reset barrier+accums
  sabr_persistent<<<NBLK, TPB, 0, stream>>>(p);
}
